// vanilla_recurrent_62328565400291
// MI455X (gfx1250) — matrix operand order repaired and checked by reading, not compiled, not run on hardware
//
#include <hip/hip_runtime.h>
#include <cmath>

// ---------------------------------------------------------------------------
// Vanilla RNN forward for MI455X (gfx1250, wave32, WMMA).
//   B=128, T=2048, N_INP=512, N_HID=512, all f32 in/out.
//   Phase 1: convert W_xh / W_hh to transposed bf16 (Wt[n][k]) in d_ws.
//   Phase 2: x_proj = inputs @ W_xh + b  -> written in place into d_out[B,T,N]
//            (bf16 WMMA, f32 accumulate; memory-bound, ~1 GB HBM traffic,
//            ~45 us at 23.3 TB/s).
//   Phase 3: persistent per-batch-group scan: H_t = tanh(x_proj_t + H @ W_hh)
//            8 workgroups (one per 16 batch rows), H double-buffered in LDS
//            as bf16, 2048 steps with ONE intra-WGP barrier per step
//            (recurrence is independent per batch row -> no global sync).
// ---------------------------------------------------------------------------

typedef __attribute__((ext_vector_type(16))) __bf16 v16bf;
typedef __attribute__((ext_vector_type(8)))  __bf16 v8bf;
typedef __attribute__((ext_vector_type(8)))  float  v8f;
typedef __attribute__((ext_vector_type(4)))  float  v4f;

#define B_SZ  128
#define T_SZ  2048
#define NINP  512
#define NHID  512
#define LDP   520            // padded LDS row stride (bf16): 1040 B, 16B aligned
#define HBUF  (16 * LDP)     // one H buffer: 16 rows

__device__ __forceinline__ v8f wmma_bf16(v16bf a, v16bf b, v8f c) {
  // D = A(16x32 bf16) * B(32x16 bf16) + C(16x16 f32)
  return __builtin_amdgcn_wmma_f32_16x16x32_bf16(
      /*neg_a=*/false, a, /*neg_b=*/false, b,
      /*c_mod=*/(short)0, c, /*reuse_a=*/false, /*reuse_b=*/false);
}

// ---------------------------------------------------------------------------
// Kernel 1: transpose + convert both weight matrices to bf16.
//   Wt[n*512 + k] = (bf16) W[k*512 + n]     (coalesced on the write side)
// ---------------------------------------------------------------------------
__global__ void convert_weights_kernel(const float* __restrict__ Wxh,
                                       const float* __restrict__ Whh,
                                       __bf16* __restrict__ Wt_xh,
                                       __bf16* __restrict__ Wt_hh) {
  const int g = blockIdx.x * blockDim.x + threadIdx.x;   // 0 .. 524287
  const int k = g & (NHID - 1);
  const int n = (g >> 9) & (NHID - 1);
  if (g < NINP * NHID) {
    Wt_xh[(size_t)n * NINP + k] = (__bf16)Wxh[(size_t)k * NHID + n];
  } else {
    Wt_hh[(size_t)n * NHID + k] = (__bf16)Whh[(size_t)k * NHID + n];
  }
}

// ---------------------------------------------------------------------------
// Kernel 2: x_proj GEMM.  Grid: (B*T)/32 blocks, 1024 threads = 32 waves.
// Each block: 32 flat rows (flat row m = b*T + t), staged once into LDS as
// bf16.  Wave w owns N-tile w (16 cols) and both 16-row M-tiles (B-fragment
// reuse).  Output written to d_out[m*512 + n] (== outs[b][t][n] slot).
// ---------------------------------------------------------------------------
__global__ void __launch_bounds__(1024)
xproj_kernel(const float* __restrict__ x, const __bf16* __restrict__ Wt,
             const float* __restrict__ bias, float* __restrict__ out) {
  __shared__ __bf16 sA[32 * LDP];

  const int tid  = threadIdx.x;
  const int lane = tid & 31;
  const int wave = tid >> 5;                      // 0..31 -> N-tile
  const long long m0 = (long long)blockIdx.x * 32;

  // ---- stage 32 rows x 512 f32 -> bf16 into LDS ----
  {
    const int row = tid >> 5;                     // 0..31
    const int c0  = (tid & 31) * 16;              // 0..496
    const float* gp = x + (m0 + row) * NINP + c0;
    __bf16* sp = sA + row * LDP + c0;
#pragma unroll
    for (int j = 0; j < 16; j += 4) {
      v4f f = *(const v4f*)(gp + j);
      sp[j + 0] = (__bf16)f.x;
      sp[j + 1] = (__bf16)f.y;
      sp[j + 2] = (__bf16)f.z;
      sp[j + 3] = (__bf16)f.w;
    }
  }
  __syncthreads();

  // ---- fragment base pointers (contiguous k = 16*kh + s in both A and B) ----
  const int n  = wave * 16 + (lane & 15);         // output column
  const int kh = lane >> 4;                       // K-half select (0/1)
  const __bf16* bb  = Wt + (size_t)n * NINP + kh * 16;  // B: 16 contig K
  const __bf16* a0p = sA + (lane & 15) * LDP + kh * 16; // A: row lane&15
  const __bf16* a1p = a0p + 16 * LDP;                   // second M-tile

  v8f acc0 = {0.f, 0.f, 0.f, 0.f, 0.f, 0.f, 0.f, 0.f};
  v8f acc1 = acc0;

#pragma unroll
  for (int k0 = 0; k0 < NINP; k0 += 32) {
    v16bf bf = *(const v16bf*)(bb + k0);          // 32B contiguous (L2-hot)
    union { v16bf v; v8bf h[2]; } ua0, ua1;
    ua0.h[0] = *(const v8bf*)(a0p + k0);
    ua0.h[1] = *(const v8bf*)(a0p + k0 + 8);
    ua1.h[0] = *(const v8bf*)(a1p + k0);
    ua1.h[1] = *(const v8bf*)(a1p + k0 + 8);
    acc0 = wmma_bf16(ua0.v, bf, acc0);            // B fragment reused
    acc1 = wmma_bf16(ua1.v, bf, acc1);
  }

  const float bv = bias[n];
  const int   mh = kh << 3;                       // row offset 0/8
#pragma unroll
  for (int r = 0; r < 8; ++r) {
    const long long m = m0 + r + mh;
    out[m * NHID + n]        = acc0[r] + bv;
    out[(m + 16) * NHID + n] = acc1[r] + bv;
  }
}

// ---------------------------------------------------------------------------
// Kernel 3: sequential scan.  Grid: 8 blocks, 1024 threads = 32 waves.
// Block g owns batch rows [16g, 16g+16).  H (16x512 bf16) is DOUBLE-BUFFERED
// in LDS: step t reads buffer (t&1), writes buffer ((t+1)&1), so a single
// barrier per step publishes H_t (no WAR hazard on the read buffer).
// Wave w owns N-tile w.  Per step: H@W_hh via 16 bf16 WMMAs (K=512), add
// x_proj (read from d_out), tanh, write H back to d_out in place + LDS.
// ---------------------------------------------------------------------------
__global__ void __launch_bounds__(1024)
rnn_scan_kernel(const float* __restrict__ H0, const __bf16* __restrict__ Wt,
                float* __restrict__ out) {
  __shared__ __bf16 sH[2 * HBUF];                 // two 16x520 bf16 buffers

  const int tid  = threadIdx.x;
  const int lane = tid & 31;
  const int wave = tid >> 5;                      // N-tile 0..31
  const int bq   = blockIdx.x * 16;               // first batch row

  // ---- seed buffer 0 with H0 (f32 -> bf16) ----
  {
    const int row = tid >> 6;                     // 0..15
    const int c0  = (tid & 63) * 8;               // 0..504
    const float* gp = H0 + (size_t)(bq + row) * NHID + c0;
    __bf16* sp = sH + row * LDP + c0;
#pragma unroll
    for (int j = 0; j < 8; ++j) sp[j] = (__bf16)gp[j];
  }
  __syncthreads();

  const int n  = wave * 16 + (lane & 15);
  const int kh = lane >> 4;
  const int mh = kh << 3;
  const __bf16* bb = Wt + (size_t)n * NHID + kh * 16;   // W_hh B fragments
  // A-fragment base within a buffer (row lane&15, K-half kh)
  const int aoff = (lane & 15) * LDP + kh * 16;
  float* HT = out + (size_t)B_SZ * T_SZ * NHID;         // final-state slot

  for (int t = 0; t < T_SZ; ++t) {
    const __bf16* ap = sH + (t & 1) * HBUF + aoff;      // read buffer
    __bf16*       wp = sH + ((t + 1) & 1) * HBUF;       // write buffer

    // Hide next step's x_proj latency behind this step's WMMA K-loop.
    if (t + 1 < T_SZ) {
      const float* pf =
          out + ((size_t)(bq + mh) * T_SZ + (t + 1)) * NHID + n;
      __builtin_prefetch(pf, 0, 1);               // -> global_prefetch_b8
    }

    v8f acc = {0.f, 0.f, 0.f, 0.f, 0.f, 0.f, 0.f, 0.f};
#pragma unroll
    for (int k0 = 0; k0 < NHID; k0 += 32) {
      union { v16bf v; v8bf h[2]; } ua;
      ua.h[0] = *(const v8bf*)(ap + k0);          // ds_load_b128
      ua.h[1] = *(const v8bf*)(ap + k0 + 8);
      v16bf bf = *(const v16bf*)(bb + k0);        // L2-resident weights
      acc = wmma_bf16(ua.v, bf, acc);
    }

#pragma unroll
    for (int r = 0; r < 8; ++r) {
      const int m = r + mh;                       // tile row 0..15
      const size_t off = ((size_t)(bq + m) * T_SZ + t) * NHID + n;
      const float h = tanhf(out[off] + acc[r]);   // x_proj + H@W_hh
      out[off] = h;                               // outs[b][t][n] in place
      wp[m * LDP + n] = (__bf16)h;                // publish H_t (other buffer)
      if (t == T_SZ - 1) HT[(size_t)(bq + m) * NHID + n] = h;
    }

    __syncthreads();   // single barrier: H_t visible to every wave
  }
}

// ---------------------------------------------------------------------------
// Host launcher.  Inputs: [inputs, H0, W_xh, W_hh, b_h]; output f32:
// outs[B,T,NHID] followed by H_T[B,NHID].
// ---------------------------------------------------------------------------
extern "C" void kernel_launch(void* const* d_in, const int* in_sizes, int n_in,
                              void* d_out, int out_size, void* d_ws, size_t ws_size,
                              hipStream_t stream) {
  const float* inputs = (const float*)d_in[0];
  const float* H0     = (const float*)d_in[1];
  const float* W_xh   = (const float*)d_in[2];
  const float* W_hh   = (const float*)d_in[3];
  const float* b_h    = (const float*)d_in[4];
  float* out = (float*)d_out;

  __bf16* Wt_xh = (__bf16*)d_ws;                  // 512 KB
  __bf16* Wt_hh = Wt_xh + (size_t)NINP * NHID;    // 512 KB

  // 1) weights -> transposed bf16 (tiny)
  convert_weights_kernel<<<2048, 256, 0, stream>>>(W_xh, W_hh, Wt_xh, Wt_hh);

  // 2) input projection GEMM: (B*T) rows / 32 per block
  xproj_kernel<<<(B_SZ * T_SZ) / 32, 1024, 0, stream>>>(inputs, Wt_xh, b_h, out);

  // 3) persistent scan: one workgroup per 16 batch rows
  rnn_scan_kernel<<<B_SZ / 16, 1024, 0, stream>>>(H0, Wt_hh, out);
}
